// MGCN_CL_52158082842662
// MI455X (gfx1250) — matrix operand order repaired and checked by reading, not compiled, not run on hardware
//
#include <hip/hip_runtime.h>
#include <hip/hip_bf16.h>
#include <math.h>

#define NN   32768
#define EE   262144
#define DD   256
#define HH   256
#define FDIM 768
#define GG   64
#define NPG  512
#define KTOP 256

typedef __attribute__((ext_vector_type(16))) __bf16 bhalf16;
typedef __attribute__((ext_vector_type(8)))  float  v8f;
typedef unsigned int u32x4 __attribute__((ext_vector_type(4)));
typedef int          i32x4 __attribute__((ext_vector_type(4)));
typedef int          i32x8 __attribute__((ext_vector_type(8)));

union AB16 { bhalf16 v; uint4 q[2]; };

__device__ __forceinline__ __bf16 f2bf(float f) {
  union { float f; unsigned u; } a; a.f = f;
  unsigned r = a.u + 0x7fffu + ((a.u >> 16) & 1u);
  unsigned short h = (unsigned short)(r >> 16);
  union { unsigned short s; __bf16 b; } o; o.s = h; return o.b;
}
__device__ __forceinline__ float bf2f(__bf16 b) {
  union { unsigned short s; __bf16 b; } i; i.b = b;
  union { unsigned u; float f; } o; o.u = ((unsigned)i.s) << 16; return o.f;
}
__device__ __forceinline__ float sigmoidf(float z) { return 1.f / (1.f + expf(-z)); }

// ---------------- WMMA GEMM with TDM-staged B tiles -------------------------------------
// C[M,Ncols] = A[M,Kd](bf16,row-major) * Bt[Ncols,Kd](bf16,row-major per out-col) (+bias)
// block = 128 threads (4 waves). Wave w: rows rowBase..rowBase+15, 64 cols.
// B tile (64 cols x 32 K) is DMA'd to LDS by the Tensor Data Mover, double-buffered.
// LDS rows padded 64B->80B stride via TDM pad fields (bank-conflict-free ds_load_b128).
#define SB_ROW   80               // bytes per LDS tile row (64 data + 16 pad)
#define SB_BUF   (64 * SB_ROW)    // 5120 B per buffer
__global__ void gemm_wmma_bf16(const __bf16* __restrict__ A, const __bf16* __restrict__ Bt,
                               float* __restrict__ C, const float* __restrict__ bias,
                               int M, int Kd, int Ncols) {
  __shared__ __align__(16) char sB[2][SB_BUF];

  const int lane = threadIdx.x & 31;
  const int wave = threadIdx.x >> 5;
  const int rowBase = blockIdx.x * 64 + wave * 16;
  const int colBase = blockIdx.y * 64;
  const int m16 = lane & 15;
  const int kA = (lane >> 4) * 8;    // A fragment K-base per half-wave
  const int kB = (lane >> 4) * 8;    // B fragment K-base per half-wave

  const int waveU = __builtin_amdgcn_readfirstlane(wave);  // SGPR copy for descriptors
  const unsigned ldsBase = (unsigned)(size_t)(&sB[0][0]);

  // ---- static part of the Tensor DMA descriptor (group 1) ----
  i32x8 g1;
  g1[0] = (int)((1u << 16)      // data_size = 1 (2-byte elements); workgroup_mask = 0
              | (1u << 20)      // pad_enable
              | (3u << 22)      // pad_interval code 3 -> every 64 bytes
              | (3u << 25));    // pad_amount  code 3 -> 16 bytes
  g1[1] = (int)((((unsigned)Kd) & 0xFFFFu) << 16);                       // tensor_dim0 lo16
  g1[2] = (int)(((((unsigned)Kd) >> 16) & 0xFFFFu) | (0xFFFFu << 16));   // dim0 hi16 | tensor_dim1 lo16
  g1[3] = (int)(32u << 16);                                              // tensor_dim1 hi16=0 | tile_dim0=32
  g1[4] = 16;                                                            // tile_dim1=16, tile_dim2=0
  g1[5] = Kd;                                                            // tensor_dim0_stride lo32
  g1[6] = (int)((((unsigned)Kd) & 0xFFFFu) << 16);                       // stride0 hi16=0 | tensor_dim1_stride lo16
  g1[7] = (int)(((unsigned)Kd) >> 16);                                   // tensor_dim1_stride hi
  const i32x4 gz4 = {0, 0, 0, 0};
  const i32x8 gz8 = {0, 0, 0, 0, 0, 0, 0, 0};

  auto tdm_issue = [&](int buf, int k0) {
    unsigned long long ga = (unsigned long long)(size_t)Bt +
        2ull * ((size_t)(colBase + waveU * 16) * (size_t)Kd + (size_t)k0);
    u32x4 g0;
    g0[0] = 1u;                                                  // count=1, user mode
    g0[1] = ldsBase + (unsigned)buf * SB_BUF + (unsigned)waveU * (16u * SB_ROW);
    g0[2] = (unsigned)ga;                                        // global_addr lo32
    g0[3] = (unsigned)((ga >> 32) & 0x01FFFFFFull) | (2u << 30); // global_addr hi | type=2
    __builtin_amdgcn_tensor_load_to_lds(g0, g1, gz4, gz4, gz8, 0);
  };

  v8f acc[4];
  #pragma unroll
  for (int t = 0; t < 4; ++t)
    #pragma unroll
    for (int i = 0; i < 8; ++i) acc[t][i] = 0.f;

  const int nIter = Kd / 32;
  tdm_issue(0, 0);                                    // prime first buffer

  const __bf16* arow = A + (size_t)(rowBase + m16) * Kd;
  for (int it = 0; it < nIter; ++it) {
    const int k0 = it * 32;
    __builtin_amdgcn_s_wait_tensorcnt(0);             // my quarter has landed
    __syncthreads();                                  // all quarters visible; prev readers done
    if (it + 1 < nIter) tdm_issue((it + 1) & 1, k0 + 32);  // overlap DMA with WMMA

    AB16 a;
    a.q[0] = *(const uint4*)(arow + k0 + kA);
    a.q[1] = *(const uint4*)(arow + k0 + kA + 16);

    const char* sbb = &sB[it & 1][0];
    #pragma unroll
    for (int t = 0; t < 4; ++t) {
      const char* bp = sbb + (t * 16 + m16) * SB_ROW + kB * 2;
      AB16 b;
      b.q[0] = *(const uint4*)(bp);
      b.q[1] = *(const uint4*)(bp + 32);
      acc[t] = __builtin_amdgcn_wmma_f32_16x16x32_bf16(false, a.v, false, b.v,
                                                       (short)0, acc[t], false, false);
    }
  }

  const int r0 = rowBase + (lane >> 4) * 8;
  #pragma unroll
  for (int t = 0; t < 4; ++t) {
    const int col = colBase + t * 16 + m16;
    const float badd = bias ? bias[col] : 0.f;
    float* cp = C + (size_t)r0 * Ncols + col;
    #pragma unroll
    for (int i = 0; i < 8; ++i) cp[(size_t)i * Ncols] = acc[t][i] + badd;
  }
}

// ---------------- degree / dinv ----------------
__global__ void deg_init(float* deg) { int i = blockIdx.x * 256 + threadIdx.x; if (i < NN) deg[i] = 1.f; }
__global__ void deg_edge(const long long* __restrict__ ei, float* deg) {
  int e = blockIdx.x * 256 + threadIdx.x; if (e >= EE) return;
  atomicAdd(&deg[(int)ei[EE + e]], 1.f);
}
__global__ void deg_to_dinv(float* deg) { int i = blockIdx.x * 256 + threadIdx.x; if (i < NN) deg[i] = rsqrtf(deg[i]); }

// ---------------- conversions ----------------
__global__ void cvt_f32_bf16(const float* __restrict__ in, __bf16* __restrict__ out, size_t n) {
  size_t i = (size_t)blockIdx.x * 256 + threadIdx.x;
  if (i < n) out[i] = f2bf(in[i]);
}
__global__ void transpose_bf16(const float* __restrict__ W, __bf16* __restrict__ Bt, int Kd, int Ncols) {
  size_t id = (size_t)blockIdx.x * 256 + threadIdx.x;
  if (id >= (size_t)Kd * Ncols) return;
  int n = (int)(id / Kd), k = (int)(id % Kd);
  Bt[id] = f2bf(W[(size_t)k * Ncols + n]);
}

// ---------------- GCN aggregation (256-wide features) ----------------
__global__ void agg_self(const float* __restrict__ h, const float* __restrict__ dinv, float* __restrict__ agg) {
  int i = blockIdx.x, f = threadIdx.x;
  float di = dinv[i];
  agg[(size_t)i * 256 + f] = h[(size_t)i * 256 + f] * di * di;
}
__global__ void agg_edge(const float* __restrict__ h, const float* __restrict__ dinv,
                         const long long* __restrict__ ei, float* __restrict__ agg) {
  int e = blockIdx.x, f = threadIdx.x;
  int s = (int)ei[e], d = (int)ei[EE + e];
  float nrm = dinv[s] * dinv[d];
  atomicAdd(&agg[(size_t)d * 256 + f], h[(size_t)s * 256 + f] * nrm);
}
__global__ void gcn_finalize(const float* __restrict__ agg, const float* __restrict__ bias, int do_lrelu,
                             __bf16* __restrict__ out, int coff, int ostride) {
  int i = blockIdx.x, f = threadIdx.x;
  float v = agg[(size_t)i * 256 + f] + (bias ? bias[f] : 0.f);
  if (do_lrelu) v = v > 0.f ? v : 0.01f * v;
  out[(size_t)i * ostride + coff + f] = f2bf(v);
}

// ---------------- score path: GEMV(768) + scalar GCN agg + sigmoid ----------------
__global__ void gemv768(const __bf16* __restrict__ hf, const float* __restrict__ w, float* __restrict__ out) {
  int wave = threadIdx.x >> 5, lane = threadIdx.x & 31;
  int row = blockIdx.x * 8 + wave;
  const __bf16* hr = hf + (size_t)row * FDIM;
  float s = 0.f;
  for (int k = lane; k < FDIM; k += 32) s += bf2f(hr[k]) * w[k];
  #pragma unroll
  for (int off = 16; off > 0; off >>= 1) s += __shfl_down(s, off, 32);
  if (lane == 0) out[row] = s;
}
__global__ void score_self(const float* __restrict__ hv, const float* __restrict__ dinv,
                           const float* __restrict__ b, float* __restrict__ spre) {
  int i = blockIdx.x * 256 + threadIdx.x;
  if (i < NN) spre[i] = hv[i] * dinv[i] * dinv[i] + b[0];
}
__global__ void score_edge(const float* __restrict__ hv, const float* __restrict__ dinv,
                           const long long* __restrict__ ei, float* __restrict__ spre) {
  int e = blockIdx.x * 256 + threadIdx.x; if (e >= EE) return;
  int s = (int)ei[e], d = (int)ei[EE + e];
  atomicAdd(&spre[d], hv[s] * dinv[s] * dinv[d]);
}
__global__ void sigmoid_store(const float* __restrict__ in, float* __restrict__ out, size_t n) {
  size_t i = (size_t)blockIdx.x * 256 + threadIdx.x;
  if (i < n) out[i] = sigmoidf(in[i]);
}
__global__ void sigmoid_inplace(float* __restrict__ a, size_t n) {
  size_t i = (size_t)blockIdx.x * 256 + threadIdx.x;
  if (i < n) a[i] = sigmoidf(a[i]);
}

// ---------------- top-k mask (K = 256 of 512, stable tie-break) ----------------
__global__ void topk_mask(const float* __restrict__ scores, int* __restrict__ mask) {
  __shared__ float s[NPG];
  int g = blockIdx.x, i = threadIdx.x;
  float si = scores[g * NPG + i];
  s[i] = si;
  __syncthreads();
  int cnt = 0;
  for (int j = 0; j < NPG; ++j) {
    float sj = s[j];
    cnt += (sj > si) || (sj == si && j < i);
  }
  mask[g * NPG + i] = (cnt < KTOP) ? 1 : 0;
}

// ---------------- readout: masked mean (/K) and masked max ----------------
__global__ void readout(const float* __restrict__ xt, const int* __restrict__ mask, float* __restrict__ e) {
  int g = blockIdx.x;
  int f = blockIdx.y * 128 + threadIdx.x;
  const float* base = xt + (size_t)g * NPG * FDIM + f;
  const int* mb = mask + g * NPG;
  float sum = 0.f, mx = -INFINITY;
  for (int n = 0; n < NPG; ++n) {
    float v = base[(size_t)n * FDIM];
    if (mb[n]) { sum += v; mx = fmaxf(mx, v); }
  }
  e[(size_t)g * (2 * FDIM) + f] = sum * (1.f / (float)KTOP);
  e[(size_t)g * (2 * FDIM) + FDIM + f] = mx;
}

// ---------------- batchnorm over 64 rows x 256 cols ----------------
__global__ void bn_stats(const float* __restrict__ z, float* __restrict__ mean, float* __restrict__ rstd) {
  int c = threadIdx.x;
  float s = 0.f, s2 = 0.f;
  for (int r = 0; r < GG; ++r) { float v = z[r * 256 + c]; s += v; s2 += v * v; }
  float m = s / (float)GG;
  float var = s2 / (float)GG - m * m;
  mean[c] = m;
  rstd[c] = rsqrtf(var + 1e-5f);
}
__global__ void bn_relu(const float* __restrict__ zg, const float* __restrict__ mean, const float* __restrict__ rstd,
                        const float* __restrict__ gamma, const float* __restrict__ beta,
                        const float* __restrict__ res, float* __restrict__ out, __bf16* __restrict__ outbf) {
  int r = blockIdx.x, c = threadIdx.x;
  float v = (zg[r * 256 + c] - mean[c]) * rstd[c] * gamma[c] + beta[c];
  v = v > 0.f ? v : 0.f;
  if (res) v += res[r * 256 + c];
  out[r * 256 + c] = v;
  outbf[r * 256 + c] = f2bf(v);
}

// ---------------- final 256 -> 10 + softmax ----------------
__global__ void final_linear(const float* __restrict__ z, const float* __restrict__ w4,
                             const float* __restrict__ b4, float* __restrict__ logits) {
  int t = threadIdx.x; if (t >= 640) return;
  int r = t / 10, c = t % 10;
  float s = b4[c];
  for (int k = 0; k < 256; ++k) s += z[r * 256 + k] * w4[k * 10 + c];
  logits[r * 10 + c] = s;
}
__global__ void softmax10(const float* __restrict__ logits, float* __restrict__ probs) {
  int r = threadIdx.x; if (r >= GG) return;
  float mx = -INFINITY;
  for (int c = 0; c < 10; ++c) mx = fmaxf(mx, logits[r * 10 + c]);
  float e[10]; float s = 0.f;
  for (int c = 0; c < 10; ++c) { e[c] = expf(logits[r * 10 + c] - mx); s += e[c]; }
  for (int c = 0; c < 10; ++c) probs[r * 10 + c] = e[c] / s;
}

__global__ void batch_out(const long long* __restrict__ batch, float* __restrict__ out) {
  int i = blockIdx.x * 256 + threadIdx.x;
  if (i < NN) out[i] = (float)batch[i];
}

// =======================================================================================
extern "C" void kernel_launch(void* const* d_in, const int* in_sizes, int n_in,
                              void* d_out, int out_size, void* d_ws, size_t ws_size,
                              hipStream_t stream) {
  const float*     x      = (const float*)d_in[0];
  const long long* ei     = (const long long*)d_in[1];
  const long long* batch  = (const long long*)d_in[2];
  const float* w_s1 = (const float*)d_in[3],  *b_s1 = (const float*)d_in[4];
  const float* w_s21= (const float*)d_in[5],  *b_s21= (const float*)d_in[6];
  const float* w_s22= (const float*)d_in[7],  *b_s22= (const float*)d_in[8];
  const float* w_raw= (const float*)d_in[9],  *b_raw= (const float*)d_in[10];
  const float* w_pos= (const float*)d_in[11], *b_pos= (const float*)d_in[12];
  const float* w_neg= (const float*)d_in[13], *b_neg= (const float*)d_in[14];
  const float* w_pool=(const float*)d_in[15], *b_pool=(const float*)d_in[16];
  const float* w1 = (const float*)d_in[17], *b1 = (const float*)d_in[18];
  const float* g1 = (const float*)d_in[19], *be1= (const float*)d_in[20];
  const float* w2 = (const float*)d_in[21], *b2 = (const float*)d_in[22];
  const float* g2 = (const float*)d_in[23], *be2= (const float*)d_in[24];
  const float* w3 = (const float*)d_in[25], *b3 = (const float*)d_in[26];
  const float* g3 = (const float*)d_in[27], *be3= (const float*)d_in[28];
  const float* w4 = (const float*)d_in[29], *b4 = (const float*)d_in[30];
  float* out = (float*)d_out;

  // output layout
  float* out_logits_pos = out;                 // 640
  float* out_probs_pos  = out + 640;           // 640
  float* out_probs_neg  = out + 1280;          // 640
  float* out_score_pos  = out + 1920;          // 32768
  float* out_score_neg  = out + 1920 + NN;     // 32768
  float* out_batch      = out + 1920 + 2*NN;   // 32768

  // workspace carve-out (256B aligned)
  char* wp = (char*)d_ws;
  auto carve = [&](size_t bytes) -> void* {
    void* p = (void*)wp; wp += (bytes + 255) & ~(size_t)255; return p;
  };
  float*  dinv   = (float*)carve(NN * 4);
  float*  svec   = (float*)carve(NN * 4);
  float*  spre   = (float*)carve(NN * 4);
  float*  meanb  = (float*)carve(256 * 4);
  float*  rstdb  = (float*)carve(256 * 4);
  float*  logtmp = (float*)carve(640 * 4);
  float*  epos   = (float*)carve((size_t)GG * 2 * FDIM * 4);
  float*  eneg   = (float*)carve((size_t)GG * 2 * FDIM * 4);
  float*  zg     = (float*)carve((size_t)GG * 256 * 4);
  float*  za     = (float*)carve((size_t)GG * 256 * 4);
  float*  zb     = (float*)carve((size_t)GG * 256 * 4);
  float*  zc     = (float*)carve((size_t)GG * 256 * 4);
  __bf16* zbf    = (__bf16*)carve((size_t)GG * 256 * 2);
  __bf16* ebf    = (__bf16*)carve((size_t)GG * 2 * FDIM * 2);
  int*    maskp  = (int*)carve(NN * 4);
  int*    maskn  = (int*)carve(NN * 4);
  __bf16* xbf    = (__bf16*)carve((size_t)NN * DD * 2);
  __bf16* hbf    = (__bf16*)carve((size_t)NN * HH * 2);
  __bf16* hfbf   = (__bf16*)carve((size_t)NN * FDIM * 2);
  __bf16* ws1t   = (__bf16*)carve((size_t)DD * HH * 2);
  __bf16* ws21t  = (__bf16*)carve((size_t)DD * HH * 2);
  __bf16* ws22t  = (__bf16*)carve((size_t)HH * HH * 2);
  __bf16* wrawt  = (__bf16*)carve((size_t)DD * HH * 2);
  __bf16* wpoolt = (__bf16*)carve((size_t)FDIM * FDIM * 2);
  __bf16* w1t    = (__bf16*)carve((size_t)2 * FDIM * 256 * 2);
  __bf16* w2t    = (__bf16*)carve((size_t)256 * 256 * 2);
  __bf16* w3t    = (__bf16*)carve((size_t)256 * 256 * 2);
  float*  hbuf   = (float*)carve((size_t)NN * 256 * 4);
  float*  aggbuf = (float*)carve((size_t)NN * 256 * 4);
  float*  xt     = (float*)carve((size_t)NN * FDIM * 4);

  const int NB = NN / 256;   // 128
  const int EB = EE / 256;   // 1024

  // degrees -> dinv
  deg_init<<<NB, 256, 0, stream>>>(dinv);
  deg_edge<<<EB, 256, 0, stream>>>(ei, dinv);
  deg_to_dinv<<<NB, 256, 0, stream>>>(dinv);

  // conversions
  cvt_f32_bf16<<<(NN * DD + 255) / 256, 256, 0, stream>>>(x, xbf, (size_t)NN * DD);
  transpose_bf16<<<(DD*HH+255)/256, 256, 0, stream>>>(w_s1,  ws1t,  DD, HH);
  transpose_bf16<<<(DD*HH+255)/256, 256, 0, stream>>>(w_s21, ws21t, DD, HH);
  transpose_bf16<<<(HH*HH+255)/256, 256, 0, stream>>>(w_s22, ws22t, HH, HH);
  transpose_bf16<<<(DD*HH+255)/256, 256, 0, stream>>>(w_raw, wrawt, DD, HH);
  transpose_bf16<<<(FDIM*FDIM+255)/256, 256, 0, stream>>>(w_pool, wpoolt, FDIM, FDIM);
  transpose_bf16<<<(2*FDIM*256+255)/256, 256, 0, stream>>>(w1, w1t, 2*FDIM, 256);
  transpose_bf16<<<(256*256+255)/256, 256, 0, stream>>>(w2, w2t, 256, 256);
  transpose_bf16<<<(256*256+255)/256, 256, 0, stream>>>(w3, w3t, 256, 256);

  dim3 gblk(128);
  dim3 gN(NN / 64, 256 / 64);      // (512,4)
  dim3 gPool(NN / 64, FDIM / 64);  // (512,12)
  dim3 gM1(1, 256 / 64);           // (1,4)

  // h_s1 = lrelu(gcn(x, w_s1)) -> hfinal cols [0,256)
  gemm_wmma_bf16<<<gN, gblk, 0, stream>>>(xbf, ws1t, hbuf, nullptr, NN, DD, HH);
  agg_self<<<NN, 256, 0, stream>>>(hbuf, dinv, aggbuf);
  agg_edge<<<EE, 256, 0, stream>>>(hbuf, dinv, ei, aggbuf);
  gcn_finalize<<<NN, 256, 0, stream>>>(aggbuf, b_s1, 1, hfbf, 0, FDIM);

  // inner gcn: lrelu(gcn(x, w_s21)) -> hbf
  gemm_wmma_bf16<<<gN, gblk, 0, stream>>>(xbf, ws21t, hbuf, nullptr, NN, DD, HH);
  agg_self<<<NN, 256, 0, stream>>>(hbuf, dinv, aggbuf);
  agg_edge<<<EE, 256, 0, stream>>>(hbuf, dinv, ei, aggbuf);
  gcn_finalize<<<NN, 256, 0, stream>>>(aggbuf, b_s21, 1, hbf, 0, HH);

  // h_s2 = lrelu(gcn(hbf, w_s22)) -> hfinal cols [256,512)
  gemm_wmma_bf16<<<gN, gblk, 0, stream>>>(hbf, ws22t, hbuf, nullptr, NN, HH, HH);
  agg_self<<<NN, 256, 0, stream>>>(hbuf, dinv, aggbuf);
  agg_edge<<<EE, 256, 0, stream>>>(hbuf, dinv, ei, aggbuf);
  gcn_finalize<<<NN, 256, 0, stream>>>(aggbuf, b_s22, 1, hfbf, 256, FDIM);

  // h_raw = x @ w_raw + b_raw (no lrelu) -> hfinal cols [512,768)
  gemm_wmma_bf16<<<gN, gblk, 0, stream>>>(xbf, wrawt, hbuf, nullptr, NN, DD, HH);
  gcn_finalize<<<NN, 256, 0, stream>>>(hbuf, b_raw, 0, hfbf, 512, FDIM);

  // scores (pos / neg): sigmoid(gcn(h_final, w))
  gemv768<<<NN / 8, 256, 0, stream>>>(hfbf, w_pos, svec);
  score_self<<<NB, 256, 0, stream>>>(svec, dinv, b_pos, spre);
  score_edge<<<EB, 256, 0, stream>>>(svec, dinv, ei, spre);
  sigmoid_store<<<NB, 256, 0, stream>>>(spre, out_score_pos, NN);

  gemv768<<<NN / 8, 256, 0, stream>>>(hfbf, w_neg, svec);
  score_self<<<NB, 256, 0, stream>>>(svec, dinv, b_neg, spre);
  score_edge<<<EB, 256, 0, stream>>>(svec, dinv, ei, spre);
  sigmoid_store<<<NB, 256, 0, stream>>>(spre, out_score_neg, NN);

  // xt = sigmoid(h_final @ w_pool + b_pool)
  gemm_wmma_bf16<<<gPool, gblk, 0, stream>>>(hfbf, wpoolt, xt, b_pool, NN, FDIM, FDIM);
  sigmoid_inplace<<<(unsigned)(((size_t)NN * FDIM + 255) / 256), 256, 0, stream>>>(xt, (size_t)NN * FDIM);

  // top-k masks
  topk_mask<<<GG, NPG, 0, stream>>>(out_score_pos, maskp);
  topk_mask<<<GG, NPG, 0, stream>>>(out_score_neg, maskn);

  // readouts
  dim3 gRead(GG, FDIM / 128);
  readout<<<gRead, 128, 0, stream>>>(xt, maskp, epos);
  readout<<<gRead, 128, 0, stream>>>(xt, maskn, eneg);

  // MLP for both branches
  for (int branch = 0; branch < 2; ++branch) {
    const float* e = branch == 0 ? epos : eneg;
    float* logits  = branch == 0 ? out_logits_pos : logtmp;
    float* probs   = branch == 0 ? out_probs_pos  : out_probs_neg;

    cvt_f32_bf16<<<(GG * 2 * FDIM + 255) / 256, 256, 0, stream>>>(e, ebf, (size_t)GG * 2 * FDIM);

    gemm_wmma_bf16<<<gM1, gblk, 0, stream>>>(ebf, w1t, zg, b1, GG, 2 * FDIM, 256);
    bn_stats<<<1, 256, 0, stream>>>(zg, meanb, rstdb);
    bn_relu<<<GG, 256, 0, stream>>>(zg, meanb, rstdb, g1, be1, nullptr, za, zbf);

    gemm_wmma_bf16<<<gM1, gblk, 0, stream>>>(zbf, w2t, zg, b2, GG, 256, 256);
    bn_stats<<<1, 256, 0, stream>>>(zg, meanb, rstdb);
    bn_relu<<<GG, 256, 0, stream>>>(zg, meanb, rstdb, g2, be2, za, zb, zbf);

    gemm_wmma_bf16<<<gM1, gblk, 0, stream>>>(zbf, w3t, zg, b3, GG, 256, 256);
    bn_stats<<<1, 256, 0, stream>>>(zg, meanb, rstdb);
    bn_relu<<<GG, 256, 0, stream>>>(zg, meanb, rstdb, g3, be3, zb, zc, zbf);

    final_linear<<<1, 640, 0, stream>>>(zc, w4, b4, logits);
    softmax10<<<1, 64, 0, stream>>>(logits, probs);
  }

  // batch passthrough
  batch_out<<<NB, 256, 0, stream>>>(batch, out_batch);
}
